// DGCNN_3212635537965
// MI455X (gfx1250) — matrix operand order repaired and checked by reading, not compiled, not run on hardware
//
#include <hip/hip_runtime.h>
#include <cstdint>
#include <cstddef>

typedef _Float16 half_t;
typedef __attribute__((ext_vector_type(16))) _Float16 v16h;
typedef __attribute__((ext_vector_type(8)))  _Float16 h8;
typedef __attribute__((ext_vector_type(8)))  float    v8f;

#define B_    8
#define N_    2048
#define KNN_  20
#define EPS_  1e-5f
#define SLOPE_ 0.2f
#define NEG_INF_ (-3.4e38f)

__device__ __forceinline__ float leaky_(float x) { return x > 0.f ? x : SLOPE_ * x; }

__device__ __forceinline__ v16h cat8_(h8 a, h8 b) {
    return __builtin_shufflevector(a, b, 0,1,2,3,4,5,6,7,8,9,10,11,12,13,14,15);
}

__device__ __forceinline__ v8f wmma_f16_(v16h a, v16h b, v8f c) {
    // 8 args: (neg_a, A, neg_b, B, c_mod, C, reuse_a, reuse_b)
    return __builtin_amdgcn_wmma_f32_16x16x32_f16(false, a, false, b, (short)0, c, false, false);
}

// ---------------------------------------------------------------------------
// Staging kernels (f32 -> f16, with padding / concat-layout remapping)
// ---------------------------------------------------------------------------

// x (8,3,2048) f32 -> xh1 (8,2048,32) f16, channels 3..31 zero-padded
__global__ void k_stage_xh1(const float* __restrict__ x, half_t* __restrict__ xh1) {
    int t = blockIdx.x * 256 + threadIdx.x;
    if (t >= B_ * N_ * 32) return;
    int c = t & 31;
    int n = (t >> 5) & (N_ - 1);
    int b = t >> 16;  // 32*2048 = 65536
    float v = (c < 3) ? x[((size_t)b * 3 + c) * N_ + n] : 0.f;
    xh1[t] = (half_t)v;
}

// edge-conv weight (Cout, 2*Cin) f32 -> (Cout, 2*Cp) f16 with pad remap
__global__ void k_stage_w_edge(const float* __restrict__ w, half_t* __restrict__ o,
                               int Cout, int Cin, int Cp) {
    int t = blockIdx.x * 256 + threadIdx.x;
    int cp2 = 2 * Cp;
    if (t >= Cout * cp2) return;
    int oo = t / cp2;
    int c = t - oo * cp2;
    float v = 0.f;
    if (c < Cp) { if (c < Cin) v = w[(size_t)oo * 2 * Cin + c]; }
    else { int c2 = c - Cp; if (c2 < Cin) v = w[(size_t)oo * 2 * Cin + Cin + c2]; }
    o[t] = (half_t)v;
}

__global__ void k_stage_mat(const float* __restrict__ w, half_t* __restrict__ o, int total) {
    int t = blockIdx.x * 256 + threadIdx.x;
    if (t >= total) return;
    o[t] = (half_t)w[t];
}

// lin3_w (40,256) -> (48,256) f16, rows 40..47 zero
__global__ void k_stage_lin3(const float* __restrict__ w, half_t* __restrict__ o) {
    int t = blockIdx.x * 256 + threadIdx.x;
    if (t >= 48 * 256) return;
    int r = t >> 8, c = t & 255;
    o[t] = (r < 40) ? (half_t)w[r * 256 + c] : (half_t)0.f;
}

// per-point squared norms from f16 features (row stride S, C real channels)
__global__ void k_norms(const half_t* __restrict__ P, int S, int C, float* __restrict__ xx) {
    int t = blockIdx.x * 256 + threadIdx.x;
    if (t >= B_ * N_) return;
    const half_t* r = P + (size_t)t * S;
    float s = 0.f;
    for (int c = 0; c < C; ++c) { float v = (float)r[c]; s += v * v; }
    xx[t] = s;
}

// ---------------------------------------------------------------------------
// KNN: neg-dist matrix via WMMA (one batch), then per-point top-20
// ---------------------------------------------------------------------------
__global__ __launch_bounds__(32) void k_knn_dist(const half_t* __restrict__ Pb, int S, int C,
                                                 const float* __restrict__ xxb,
                                                 float* __restrict__ D) {
    const int lane = threadIdx.x;
    const int nt = blockIdx.x, mt = blockIdx.y;
    const int l16 = lane & 15;
    const int kbase = (lane < 16) ? 0 : 8;
    const int koff  = (lane < 16) ? 0 : 16;
    const half_t* ar = Pb + (size_t)(nt * 16 + l16) * S;
    const half_t* br = Pb + (size_t)(mt * 16 + l16) * S;
    v8f acc = {};
    for (int kc = 0; kc < C; kc += 32) {
        h8 r0 = *(const h8*)(ar + kc + kbase);
        h8 r1 = *(const h8*)(ar + kc + kbase + 16);
        v16h a = cat8_(r0, r1);
        v16h bf = cat8_(*(const h8*)(br + kc + kbase), *(const h8*)(br + kc + kbase + 16));
        acc = wmma_f16_(a, bf, acc);
    }
    int m = mt * 16 + l16;
    float xxm = xxb[m];
    int m0 = (lane < 16) ? 0 : 8;
#pragma unroll
    for (int v = 0; v < 8; ++v) {
        int n = nt * 16 + m0 + v;
        D[(size_t)n * N_ + m] = 2.f * acc[v] - xxb[n] - xxm;
    }
}

__global__ __launch_bounds__(256) void k_topk(const float* __restrict__ D,
                                              int* __restrict__ idxb) {
    int n = blockIdx.x * 256 + threadIdx.x;
    float bd[KNN_]; int bi[KNN_];
#pragma unroll
    for (int i = 0; i < KNN_; ++i) { bd[i] = NEG_INF_; bi[i] = 0; }
    for (int m = 0; m < N_; ++m) {
        float d = D[(size_t)m * N_ + n];  // D symmetric: column scan = coalesced
        if (d > bd[KNN_ - 1]) {
            float dv = d; int key = m;
#pragma unroll
            for (int i = 0; i < KNN_; ++i) {
                if (dv > bd[i]) {
                    float tf = bd[i]; bd[i] = dv; dv = tf;
                    int ti = bi[i]; bi[i] = key; key = ti;
                }
            }
        }
    }
#pragma unroll
    for (int i = 0; i < KNN_; ++i) idxb[(size_t)n * KNN_ + i] = bi[i];
}

// ---------------------------------------------------------------------------
// Fused edge-conv: gather -> WMMA GEMM -> BN -> leaky -> max over k
// One wave per (16 points x 16 out-channels). Rows are point-major (p*20+kk).
// ---------------------------------------------------------------------------
template <int CP>
__global__ __launch_bounds__(32) void k_edgeconv(
    const half_t* __restrict__ P, int S, const int* __restrict__ idx,
    const half_t* __restrict__ wh,
    const float* __restrict__ bg, const float* __restrict__ bb,
    const float* __restrict__ bm, const float* __restrict__ bv,
    half_t* __restrict__ outP, int outS) {
    constexpr int K2 = 2 * CP;
    __shared__ alignas(128) half_t wt[16 * K2];
    __shared__ float hbuf[320 * 16];
    const int lane = threadIdx.x;
    const int ntile = blockIdx.x, otile = blockIdx.y, b = blockIdx.z;

    // stage 16 x K2 weight tile into LDS (vectorized)
    for (int t = lane; t < 16 * (K2 / 8); t += 32) {
        int r = t / (K2 / 8), c8 = (t % (K2 / 8)) * 8;
        *(h8*)&wt[r * K2 + c8] = *(const h8*)&wh[(size_t)(otile * 16 + r) * K2 + c8];
    }
    __syncthreads();

    const int m_ = lane & 15;
    const int kbase = (lane < 16) ? 0 : 8;
    const int koff  = (lane < 16) ? 0 : 16;

    for (int rt = 0; rt < 20; ++rt) {
        int R = rt * 16 + m_;
        int pl = R / 20, kk = R - pl * 20;
        int pg = ntile * 16 + pl;
        int j = idx[((size_t)b * N_ + pg) * KNN_ + kk];
        const half_t* rp = P + ((size_t)b * N_ + pg) * S;
        const half_t* rj = P + ((size_t)b * N_ + j) * S;
        v8f acc = {};
#pragma unroll
        for (int kc = 0; kc < K2; kc += 32) {
            int cb0 = kc + kbase, cb1 = kc + kbase + 16;
            h8 r0 = (cb0 < CP) ? h8(*(const h8*)(rj + cb0) - *(const h8*)(rp + cb0))
                               : *(const h8*)(rp + (cb0 - CP));
            h8 r1 = (cb1 < CP) ? h8(*(const h8*)(rj + cb1) - *(const h8*)(rp + cb1))
                               : *(const h8*)(rp + (cb1 - CP));
            v16h a = cat8_(r0, r1);
            v16h bf = cat8_(*(const h8*)&wt[m_ * K2 + kc + kbase], *(const h8*)&wt[m_ * K2 + kc + kbase + 16]);
            acc = wmma_f16_(a, bf, acc);
        }
        int m0 = (lane < 16) ? 0 : 8;
#pragma unroll
        for (int v = 0; v < 8; ++v)
            hbuf[(rt * 16 + m0 + v) * 16 + m_] = acc[v];
    }
    __syncthreads();

    // BN + leaky + max over 20 neighbors; write f16 into concat buffer
    const int o_l = lane & 15;
    const int og = otile * 16 + o_l;
    const float s = bg[og] * rsqrtf(bv[og] + EPS_);
    const float t = bb[og] - bm[og] * s;
#pragma unroll
    for (int q = 0; q < 8; ++q) {
        int p_l = (lane >> 4) + 2 * q;
        float best = NEG_INF_;
        for (int kk = 0; kk < KNN_; ++kk) {
            float hv = hbuf[(p_l * KNN_ + kk) * 16 + o_l] * s + t;
            best = fmaxf(best, leaky_(hv));
        }
        outP[((size_t)b * N_ + ntile * 16 + p_l) * outS + og] = (half_t)best;
    }
}

// ---------------------------------------------------------------------------
// w5 GEMM (B*N x 512 -> 1024) + BN5 + leaky, fused deterministic pooling
// partials: per (b, n-tile) max and sum over 16 rows
// ---------------------------------------------------------------------------
__global__ __launch_bounds__(32) void k_w5pool(
    const half_t* __restrict__ xcat, const half_t* __restrict__ w5h,
    const float* __restrict__ bg, const float* __restrict__ bb,
    const float* __restrict__ bm, const float* __restrict__ bv,
    float* __restrict__ pmax, float* __restrict__ psum) {
    const int lane = threadIdx.x;
    const int nt = blockIdx.x, ot = blockIdx.y, b = blockIdx.z;
    const int l16 = lane & 15;
    const int kbase = (lane < 16) ? 0 : 8;
    const int koff  = (lane < 16) ? 0 : 16;
    const half_t* ar = xcat + ((size_t)b * N_ + nt * 16 + l16) * 512;
    const half_t* br = w5h + (size_t)(ot * 16 + l16) * 512;
    v8f acc = {};
#pragma unroll 4
    for (int kc = 0; kc < 512; kc += 32) {
        h8 r0 = *(const h8*)(ar + kc + kbase);
        h8 r1 = *(const h8*)(ar + kc + kbase + 16);
        v16h a = cat8_(r0, r1);
        v16h bf = cat8_(*(const h8*)(br + kc + kbase), *(const h8*)(br + kc + kbase + 16));
        acc = wmma_f16_(a, bf, acc);
    }
    int o = ot * 16 + l16;
    float s = bg[o] * rsqrtf(bv[o] + EPS_);
    float t = bb[o] - bm[o] * s;
    float cmax = NEG_INF_, csum = 0.f;
#pragma unroll
    for (int v = 0; v < 8; ++v) {
        float val = leaky_(acc[v] * s + t);
        cmax = fmaxf(cmax, val);
        csum += val;
    }
    cmax = fmaxf(cmax, __shfl_xor(cmax, 16, 32));
    csum = csum + __shfl_xor(csum, 16, 32);
    if (lane < 16) {
        size_t base = ((size_t)b * 128 + nt) * 1024 + o;
        pmax[base] = cmax;
        psum[base] = csum;
    }
}

// fixed-order reduction of partials -> g = [max, mean] as f16 (8 x 2048)
__global__ void k_pool_reduce(const float* __restrict__ pmax, const float* __restrict__ psum,
                              half_t* __restrict__ gh) {
    int t = blockIdx.x * 256 + threadIdx.x;
    if (t >= B_ * 1024) return;
    int b = t >> 10, o = t & 1023;
    float mx = NEG_INF_, sm = 0.f;
    for (int nt = 0; nt < 128; ++nt) {
        size_t base = ((size_t)b * 128 + nt) * 1024 + o;
        mx = fmaxf(mx, pmax[base]);
        sm += psum[base];
    }
    gh[(size_t)b * 2048 + o] = (half_t)mx;
    gh[(size_t)b * 2048 + 1024 + o] = (half_t)(sm * (1.f / 2048.f));
}

// ---------------------------------------------------------------------------
// Small-M (8-row) FC layer via one WMMA tile per 16 output columns
// ---------------------------------------------------------------------------
template <bool BIAS, bool BN, bool LEAKY>
__global__ __launch_bounds__(32) void k_smallgemm(
    const half_t* __restrict__ A, const half_t* __restrict__ W,
    const float* __restrict__ bias,
    const float* __restrict__ bg, const float* __restrict__ bb,
    const float* __restrict__ bm, const float* __restrict__ bv,
    float* __restrict__ outf, half_t* __restrict__ outh,
    int K, int Nreal, int outStride) {
    const int lane = threadIdx.x;
    const int tile = blockIdx.x;
    const int l16 = lane & 15;
    const int kbase = (lane < 16) ? 0 : 8;
    const int koff  = (lane < 16) ? 0 : 16;
    const bool valid = l16 < 8;
    const half_t* ar = A + (size_t)l16 * K;
    const half_t* br = W + (size_t)(tile * 16 + l16) * K;
    v8f acc = {};
    for (int kc = 0; kc < K; kc += 32) {
        h8 z = {};
        h8 r0 = valid ? *(const h8*)(ar + kc + kbase) : z;
        h8 r1 = valid ? *(const h8*)(ar + kc + kbase + 16) : z;
        v16h a = cat8_(r0, r1);
        v16h bf = cat8_(*(const h8*)(br + kc + kbase), *(const h8*)(br + kc + kbase + 16));
        acc = wmma_f16_(a, bf, acc);
    }
    if (lane < 16) {
        int o = tile * 16 + lane;
        if (o < Nreal) {
            float s = 1.f, t = 0.f, bi = 0.f;
            if (BN) { s = bg[o] * rsqrtf(bv[o] + EPS_); t = bb[o] - bm[o] * s; }
            if (BIAS) bi = bias[o];
#pragma unroll
            for (int v = 0; v < 8; ++v) {  // row v = batch index
                float val = acc[v] + bi;
                if (BN) val = val * s + t;
                if (LEAKY) val = leaky_(val);
                if (outh) outh[(size_t)v * outStride + o] = (half_t)val;
                if (outf) outf[(size_t)v * outStride + o] = val;
            }
        }
    }
}

// ---------------------------------------------------------------------------
// Launcher
// ---------------------------------------------------------------------------
extern "C" void kernel_launch(void* const* d_in, const int* in_sizes, int n_in,
                              void* d_out, int out_size, void* d_ws, size_t ws_size,
                              hipStream_t stream) {
    (void)in_sizes; (void)n_in; (void)out_size; (void)ws_size;
    const float* x    = (const float*)d_in[0];
    const float* w1   = (const float*)d_in[1];
    const float *bn1g = (const float*)d_in[2],  *bn1b = (const float*)d_in[3],
                *bn1m = (const float*)d_in[4],  *bn1v = (const float*)d_in[5];
    const float* w2   = (const float*)d_in[6];
    const float *bn2g = (const float*)d_in[7],  *bn2b = (const float*)d_in[8],
                *bn2m = (const float*)d_in[9],  *bn2v = (const float*)d_in[10];
    const float* w3   = (const float*)d_in[11];
    const float *bn3g = (const float*)d_in[12], *bn3b = (const float*)d_in[13],
                *bn3m = (const float*)d_in[14], *bn3v = (const float*)d_in[15];
    const float* w4   = (const float*)d_in[16];
    const float *bn4g = (const float*)d_in[17], *bn4b = (const float*)d_in[18],
                *bn4m = (const float*)d_in[19], *bn4v = (const float*)d_in[20];
    const float* w5   = (const float*)d_in[21];
    const float *bn5g = (const float*)d_in[22], *bn5b = (const float*)d_in[23],
                *bn5m = (const float*)d_in[24], *bn5v = (const float*)d_in[25];
    const float* l1w  = (const float*)d_in[26];
    const float *bn6g = (const float*)d_in[27], *bn6b = (const float*)d_in[28],
                *bn6m = (const float*)d_in[29], *bn6v = (const float*)d_in[30];
    const float* l2w  = (const float*)d_in[31];
    const float* l2b  = (const float*)d_in[32];
    const float *bn7g = (const float*)d_in[33], *bn7b = (const float*)d_in[34],
                *bn7m = (const float*)d_in[35], *bn7v = (const float*)d_in[36];
    const float* l3w  = (const float*)d_in[37];
    const float* l3b  = (const float*)d_in[38];
    float* out = (float*)d_out;

    char* ws = (char*)d_ws;
    size_t off = 0;
    auto alloc = [&](size_t bytes) {
        size_t o = off;
        off += (bytes + 255) & ~(size_t)255;
        return o;
    };
    half_t* xh1  = (half_t*)(ws + alloc((size_t)B_ * N_ * 32 * 2));
    half_t* xcat = (half_t*)(ws + alloc((size_t)B_ * N_ * 512 * 2));
    half_t* w1h  = (half_t*)(ws + alloc((size_t)64 * 64 * 2));
    half_t* w2h  = (half_t*)(ws + alloc((size_t)64 * 128 * 2));
    half_t* w3h  = (half_t*)(ws + alloc((size_t)128 * 128 * 2));
    half_t* w4h  = (half_t*)(ws + alloc((size_t)256 * 256 * 2));
    half_t* w5h  = (half_t*)(ws + alloc((size_t)1024 * 512 * 2));
    half_t* l1h  = (half_t*)(ws + alloc((size_t)512 * 2048 * 2));
    half_t* l2h  = (half_t*)(ws + alloc((size_t)256 * 512 * 2));
    half_t* l3h  = (half_t*)(ws + alloc((size_t)48 * 256 * 2));
    float*  xx   = (float*)(ws + alloc((size_t)B_ * N_ * 4));
    float*  D    = (float*)(ws + alloc((size_t)N_ * N_ * 4));
    int*    idx  = (int*)(ws + alloc((size_t)B_ * N_ * KNN_ * 4));
    float*  pmax = (float*)(ws + alloc((size_t)B_ * 128 * 1024 * 4));
    float*  psum = (float*)(ws + alloc((size_t)B_ * 128 * 1024 * 4));
    half_t* gh   = (half_t*)(ws + alloc((size_t)B_ * 2048 * 2));
    half_t* g1h  = (half_t*)(ws + alloc((size_t)B_ * 512 * 2));
    half_t* g2h  = (half_t*)(ws + alloc((size_t)B_ * 256 * 2));

    auto blocks = [](int total) { return dim3((total + 255) / 256); };

    // ---- stage inputs / weights to f16 ----
    k_stage_xh1<<<blocks(B_ * N_ * 32), 256, 0, stream>>>(x, xh1);
    k_stage_w_edge<<<blocks(64 * 64), 256, 0, stream>>>(w1, w1h, 64, 3, 32);
    k_stage_w_edge<<<blocks(64 * 128), 256, 0, stream>>>(w2, w2h, 64, 64, 64);
    k_stage_w_edge<<<blocks(128 * 128), 256, 0, stream>>>(w3, w3h, 128, 64, 64);
    k_stage_w_edge<<<blocks(256 * 256), 256, 0, stream>>>(w4, w4h, 256, 128, 128);
    k_stage_mat<<<blocks(1024 * 512), 256, 0, stream>>>(w5, w5h, 1024 * 512);
    k_stage_mat<<<blocks(512 * 2048), 256, 0, stream>>>(l1w, l1h, 512 * 2048);
    k_stage_mat<<<blocks(256 * 512), 256, 0, stream>>>(l2w, l2h, 256 * 512);
    k_stage_lin3<<<blocks(48 * 256), 256, 0, stream>>>(l3w, l3h);

    // ---- layer 1: features xh1 (C=32 padded), Cout=64 -> xcat[:,0:64) ----
    k_norms<<<blocks(B_ * N_), 256, 0, stream>>>(xh1, 32, 32, xx);
    for (int b = 0; b < B_; ++b) {
        k_knn_dist<<<dim3(N_ / 16, N_ / 16), 32, 0, stream>>>(
            xh1 + (size_t)b * N_ * 32, 32, 32, xx + b * N_, D);
        k_topk<<<N_ / 256, 256, 0, stream>>>(D, idx + (size_t)b * N_ * KNN_);
    }
    k_edgeconv<32><<<dim3(N_ / 16, 4, B_), 32, 0, stream>>>(
        xh1, 32, idx, w1h, bn1g, bn1b, bn1m, bn1v, xcat + 0, 512);

    // ---- layer 2: x1 (C=64) -> xcat[:,64:128) ----
    k_norms<<<blocks(B_ * N_), 256, 0, stream>>>(xcat + 0, 512, 64, xx);
    for (int b = 0; b < B_; ++b) {
        k_knn_dist<<<dim3(N_ / 16, N_ / 16), 32, 0, stream>>>(
            xcat + 0 + (size_t)b * N_ * 512, 512, 64, xx + b * N_, D);
        k_topk<<<N_ / 256, 256, 0, stream>>>(D, idx + (size_t)b * N_ * KNN_);
    }
    k_edgeconv<64><<<dim3(N_ / 16, 4, B_), 32, 0, stream>>>(
        xcat + 0, 512, idx, w2h, bn2g, bn2b, bn2m, bn2v, xcat + 64, 512);

    // ---- layer 3: x2 (C=64) -> xcat[:,128:256), Cout=128 ----
    k_norms<<<blocks(B_ * N_), 256, 0, stream>>>(xcat + 64, 512, 64, xx);
    for (int b = 0; b < B_; ++b) {
        k_knn_dist<<<dim3(N_ / 16, N_ / 16), 32, 0, stream>>>(
            xcat + 64 + (size_t)b * N_ * 512, 512, 64, xx + b * N_, D);
        k_topk<<<N_ / 256, 256, 0, stream>>>(D, idx + (size_t)b * N_ * KNN_);
    }
    k_edgeconv<64><<<dim3(N_ / 16, 8, B_), 32, 0, stream>>>(
        xcat + 64, 512, idx, w3h, bn3g, bn3b, bn3m, bn3v, xcat + 128, 512);

    // ---- layer 4: x3 (C=128) -> xcat[:,256:512), Cout=256 ----
    k_norms<<<blocks(B_ * N_), 256, 0, stream>>>(xcat + 128, 512, 128, xx);
    for (int b = 0; b < B_; ++b) {
        k_knn_dist<<<dim3(N_ / 16, N_ / 16), 32, 0, stream>>>(
            xcat + 128 + (size_t)b * N_ * 512, 512, 128, xx + b * N_, D);
        k_topk<<<N_ / 256, 256, 0, stream>>>(D, idx + (size_t)b * N_ * KNN_);
    }
    k_edgeconv<128><<<dim3(N_ / 16, 16, B_), 32, 0, stream>>>(
        xcat + 128, 512, idx, w4h, bn4g, bn4b, bn4m, bn4v, xcat + 256, 512);

    // ---- w5 GEMM + BN5 + leaky + fused max/mean pooling ----
    k_w5pool<<<dim3(N_ / 16, 1024 / 16, B_), 32, 0, stream>>>(
        xcat, w5h, bn5g, bn5b, bn5m, bn5v, pmax, psum);
    k_pool_reduce<<<blocks(B_ * 1024), 256, 0, stream>>>(pmax, psum, gh);

    // ---- FC head ----
    k_smallgemm<false, true, true><<<512 / 16, 32, 0, stream>>>(
        gh, l1h, nullptr, bn6g, bn6b, bn6m, bn6v, nullptr, g1h, 2048, 512, 512);
    k_smallgemm<true, true, true><<<256 / 16, 32, 0, stream>>>(
        g1h, l2h, l2b, bn7g, bn7b, bn7m, bn7v, nullptr, g2h, 512, 256, 256);
    k_smallgemm<true, false, false><<<48 / 16, 32, 0, stream>>>(
        g2h, l3h, l3b, nullptr, nullptr, nullptr, nullptr, out, nullptr, 256, 40, 40);
}
